// Qwen3VLModelWithAllOptimizations_13932873908552
// MI455X (gfx1250) — compile-verified, matrix operand order checked by reading, not run on hardware
//
#include <hip/hip_runtime.h>
#include <hip/hip_bf16.h>

typedef __attribute__((ext_vector_type(16))) __bf16 v16bf;
typedef __attribute__((ext_vector_type(8)))  __bf16 v8bf;
typedef __attribute__((ext_vector_type(8)))  float  v8f;
typedef __attribute__((ext_vector_type(4)))  unsigned u32x4;
typedef __attribute__((ext_vector_type(8)))  unsigned u32x8;

__device__ __forceinline__ float bf2f(__bf16 b) {
    unsigned short s = __builtin_bit_cast(unsigned short, b);
    unsigned u = ((unsigned)s) << 16;
    return __builtin_bit_cast(float, u);
}
__device__ __forceinline__ __bf16 f2bf(float f) {
    unsigned u = __builtin_bit_cast(unsigned, f);
    unsigned r = u + 0x7FFFu + ((u >> 16) & 1u);
    unsigned short s = (unsigned short)(r >> 16);
    return __builtin_bit_cast(__bf16, s);
}

// For __shared__ pointers the generic address low 32 bits are the LDS offset
// (aperture selects LDS via addr[63:32]).
__device__ __forceinline__ unsigned lds_off(const void* p) {
    return (unsigned)(size_t)p;
}

// per-lane async copy global -> LDS, tracked by ASYNCcnt
__device__ __forceinline__ void async_load_b128(unsigned lds_addr, const void* gptr) {
    asm volatile("global_load_async_to_lds_b128 %0, %1, off"
                 :: "v"(lds_addr), "v"((unsigned long long)(size_t)gptr)
                 : "memory");
}
__device__ __forceinline__ void wait_async(int n) {
    if (n == 0) asm volatile("s_wait_asynccnt 0x0" ::: "memory");
    else        asm volatile("s_wait_asynccnt 0x1" ::: "memory");
}

// TDM: 2-D tile load (bf16 elements) global -> LDS via tensor descriptor.
// tile_k contiguous elements per row (dim0), tile_n rows (dim1), row stride = K elems.
__device__ __forceinline__ void tdm_load_2d(unsigned lds_addr, const void* gptr,
                                            unsigned K, unsigned N,
                                            unsigned tile_k, unsigned tile_n) {
    unsigned long long ga = (unsigned long long)(size_t)gptr;
    u32x4 g0;
    g0[0] = 1u;                                   // count=1, is_restore=0, no gather
    g0[1] = lds_addr;                             // lds_addr [63:32]
    g0[2] = (unsigned)ga;                         // global_addr [95:64]
    g0[3] = (unsigned)(ga >> 32) | (2u << 30);    // global_addr hi + type=2 [127:126]
    u32x8 g1;
    g1[0] = 0x10000u;                             // workgroup_mask=0, data_size=1 (2B)
    g1[1] = (K & 0xFFFFu) << 16;                  // tensor_dim0 lo16 at [63:48]
    g1[2] = (K >> 16) | ((N & 0xFFFFu) << 16);    // tensor_dim0 hi16, tensor_dim1 lo16
    g1[3] = (N >> 16) | (tile_k << 16);           // tensor_dim1 hi16, tile_dim0 [127:112]
    g1[4] = tile_n & 0xFFFFu;                     // tile_dim1 [143:128], tile_dim2=0
    g1[5] = K;                                    // tensor_dim0_stride lo32
    g1[6] = 0u;                                   // stride hi16, dim1_stride lo16
    g1[7] = 0u;
    asm volatile("tensor_load_to_lds %0, %1" :: "s"(g0), "s"(g1) : "memory");
}

// ---------------- elementwise kernels ----------------

__global__ void gather_kernel(const int* __restrict__ ids, const float* __restrict__ embed,
                              float* __restrict__ x, int Hd, int total) {
    int idx = blockIdx.x * 256 + threadIdx.x;
    if (idx >= total) return;
    int row = idx / Hd, c = idx % Hd;
    x[idx] = embed[(size_t)ids[row] * Hd + c];
}

// transpose+convert: in f32 [K,N] -> out bf16 [N,K]
__global__ void wtrans_kernel(const float* __restrict__ in, __bf16* __restrict__ out,
                              int K, int N) {
    int idx = blockIdx.x * 256 + threadIdx.x;
    if (idx >= K * N) return;
    int n = idx / K, k = idx % K;
    out[idx] = f2bf(in[(size_t)k * N + n]);
}

__global__ __launch_bounds__(256) void ln_kernel(const float* __restrict__ x,
                                                 const float* __restrict__ w,
                                                 const float* __restrict__ bsrc,
                                                 __bf16* __restrict__ out, int Hd) {
    int row = blockIdx.x, tid = threadIdx.x;
    __shared__ float red[256];
    const float* xp = x + (size_t)row * Hd;
    float s = 0.f;
    for (int i = tid; i < Hd; i += 256) s += xp[i];
    red[tid] = s; __syncthreads();
    for (int o = 128; o > 0; o >>= 1) { if (tid < o) red[tid] += red[tid + o]; __syncthreads(); }
    float mean = red[0] / (float)Hd;
    __syncthreads();
    float v = 0.f;
    for (int i = tid; i < Hd; i += 256) { float d = xp[i] - mean; v += d * d; }
    red[tid] = v; __syncthreads();
    for (int o = 128; o > 0; o >>= 1) { if (tid < o) red[tid] += red[tid + o]; __syncthreads(); }
    float rstd = rsqrtf(red[0] / (float)Hd + 1e-6f);
    for (int i = tid; i < Hd; i += 256)
        out[(size_t)row * Hd + i] = f2bf((xp[i] - mean) * rstd * w[i] + bsrc[i]);
}

// RoPE + layout change: f32 [B,S,Hd] -> bf16 [B,H,S,D]
__global__ void rope_kernel(const float* __restrict__ qf, const float* __restrict__ kf,
                            const float* __restrict__ vf,
                            __bf16* __restrict__ qb, __bf16* __restrict__ kb,
                            __bf16* __restrict__ vb,
                            int B, int H, int S, int D, int Hd) {
    int idx = blockIdx.x * 256 + threadIdx.x;
    if (idx >= B * H * S * D) return;
    int d = idx % D; int t = idx / D;
    int s = t % S;  int bh = t / S;
    int hh = bh % H; int b = bh / H;
    size_t src = (size_t)(b * S + s) * Hd + hh * D + d;
    int p = d & 31;
    float inv = expf(-((float)(2 * p) / (float)D) * logf(1.0e6f));
    float f = (float)s * inv;
    float c = cosf(f), sn = sinf(f);
    size_t part = (d < 32) ? src + 32 : src - 32;
    float qr = (d < 32) ? -qf[part] : qf[part];
    float kr = (d < 32) ? -kf[part] : kf[part];
    qb[idx] = f2bf(qf[src] * c + qr * sn);
    kb[idx] = f2bf(kf[src] * c + kr * sn);
    vb[idx] = f2bf(vf[src]);
}

// router[b,h,s] = dot(h_ln[b,s,:], Wr[:,h])
__global__ void router_kernel(const __bf16* __restrict__ h, const float* __restrict__ Wr,
                              float* __restrict__ router, int B, int H, int S, int Hd) {
    int idx = blockIdx.x * 256 + threadIdx.x;
    if (idx >= B * H * S) return;
    int s = idx % S; int bh = idx / S; int hh = bh % H;
    int b = bh / H;
    const __bf16* hp = h + (size_t)(b * S + s) * Hd;
    float acc = 0.f;
    for (int e = 0; e < Hd; e++) acc += bf2f(hp[e]) * Wr[e * H + hh];
    router[idx] = acc;
}

// exact kth-largest per (b,h) via radix threshold search on sortable uint keys
__global__ __launch_bounds__(256) void kth_select_kernel(const float* __restrict__ router,
                                                         float* __restrict__ kth,
                                                         int S, int topk) {
    int bh = blockIdx.x;
    __shared__ unsigned keys[1024];
    __shared__ int red[256];
    __shared__ unsigned Tsh;
    int tid = threadIdx.x;
    for (int i = tid; i < S; i += 256) {
        unsigned u = __builtin_bit_cast(unsigned, router[(size_t)bh * S + i]);
        u = (u & 0x80000000u) ? ~u : (u | 0x80000000u);
        keys[i] = u;
    }
    if (tid == 0) Tsh = 0u;
    __syncthreads();
    for (int bit = 31; bit >= 0; --bit) {
        unsigned cand = Tsh | (1u << bit);
        int c = 0;
        for (int i = tid; i < S; i += 256) c += (keys[i] >= cand) ? 1 : 0;
        red[tid] = c; __syncthreads();
        for (int o = 128; o > 0; o >>= 1) { if (tid < o) red[tid] += red[tid + o]; __syncthreads(); }
        if (tid == 0 && red[0] >= topk) Tsh = cand;
        __syncthreads();
    }
    if (tid == 0) {
        unsigned u = Tsh;
        u = (u & 0x80000000u) ? (u & 0x7FFFFFFFu) : ~u;
        kth[bh] = __builtin_bit_cast(float, u);
    }
}

// biasT[b,h,s] = (router >= kth ? log_sigmoid(router) : NEG) + (1-mask)*NEG
__global__ void bias_kernel(const float* __restrict__ router, const float* __restrict__ kth,
                            const float* __restrict__ amask, float* __restrict__ biasT,
                            int B, int H, int S) {
    int idx = blockIdx.x * 256 + threadIdx.x;
    if (idx >= B * H * S) return;
    int s = idx % S; int bh = idx / S; int b = bh / H;
    float r = router[idx];
    float ls = (r >= 0.f) ? (-log1pf(expf(-r))) : (r - log1pf(expf(r)));
    float v = (r >= kth[bh]) ? ls : -1e9f;
    v += (1.0f - amask[b * S + s]) * -1e9f;
    biasT[idx] = v;
}

// ---------------- WMMA GEMM: C[M,N] = A[M,K](bf16) x BT[N,K](bf16) ----------------
// block tile 128x64, 8 waves (4x2), wave tile 32x32 (2x2 WMMA 16x16x32).
// Double-buffered LDS; A staged with global_load_async_to_lds_b128 (ASYNCcnt),
// B staged with TDM tensor_load_to_lds (TENSORcnt, issued by wave 0).
template <int ACT> // 0 = none, 1 = SiLU
__global__ __launch_bounds__(256) void gemm_bf16_kernel(
    const __bf16* __restrict__ A, const __bf16* __restrict__ BT,
    const float* __restrict__ bias, const float* resid,
    float* Cf, __bf16* Cb, int M, int N, int K) {
    __shared__ __bf16 As[2][128 * 32];
    __shared__ __bf16 Bs[2][64 * 32];
    int tid = threadIdx.x, lane = tid & 31, wave = tid >> 5;
    int wm = wave >> 1, wn = wave & 1;
    int m0 = blockIdx.y * 128, n0 = blockIdx.x * 64;
    int lr = lane & 15, lsel = (lane >> 4) * 16;
    int row = tid >> 1, half = tid & 1;
    v8f acc[2][2] = {};

    auto issue_tile = [&](int k0, int buf) {
        async_load_b128(lds_off(&As[buf][row * 32 + half * 16]),
                        A + (size_t)(m0 + row) * K + k0 + half * 16);
        if (wave == 0)
            tdm_load_2d(lds_off(&Bs[buf][0]), BT + (size_t)n0 * K + k0,
                        (unsigned)K, (unsigned)N, 32u, 64u);
    };

    int T = K / 32;
    issue_tile(0, 0);
    for (int t = 0; t < T; ++t) {
        int buf = t & 1;
        bool more = (t + 1 < T);
        if (more) issue_tile((t + 1) * 32, buf ^ 1);
        wait_async(more ? 1 : 0);
        if (wave == 0) {
            if (more) __builtin_amdgcn_s_wait_tensorcnt((short)1);
            else      __builtin_amdgcn_s_wait_tensorcnt((short)0);
        }
        __syncthreads();
        v16bf afr[2], bfr[2];
        for (int i = 0; i < 2; i++)
            afr[i] = *(const v16bf*)&As[buf][(wm * 32 + i * 16 + lr) * 32 + lsel];
        for (int j = 0; j < 2; j++)
            bfr[j] = *(const v16bf*)&Bs[buf][(wn * 32 + j * 16 + lr) * 32 + lsel];
        for (int i = 0; i < 2; i++)
            for (int j = 0; j < 2; j++)
                acc[i][j] = __builtin_amdgcn_wmma_f32_16x16x32_bf16(
                    false, afr[i], false, bfr[j], (short)0, acc[i][j], false, false);
        __syncthreads();
    }
    for (int i = 0; i < 2; i++)
        for (int j = 0; j < 2; j++)
            for (int e = 0; e < 8; e++) {
                int r2 = m0 + wm * 32 + i * 16 + e + 8 * (lane >> 4);
                int col = n0 + wn * 32 + j * 16 + (lane & 15);
                float val = acc[i][j][e];
                if (bias) val += bias[col];
                if (ACT == 1) val = val / (1.f + expf(-val));
                if (resid) val += resid[(size_t)r2 * N + col];
                if (Cf) Cf[(size_t)r2 * N + col] = val;
                if (Cb) Cb[(size_t)r2 * N + col] = f2bf(val);
            }
}

// ---------------- flash attention with WMMA (64x64 tiles, online softmax) ----------------
__global__ __launch_bounds__(256) void flash_attn_kernel(
    const __bf16* __restrict__ qb, const __bf16* __restrict__ kb,
    const __bf16* __restrict__ vb, const float* __restrict__ biasT,
    __bf16* __restrict__ ob, int B, int H, int S, int D, int Hd) {
    const int QT = 64, KT = 64;
    __shared__ __bf16 Qs[64 * 64];
    __shared__ __bf16 Ks[64 * 64];
    __shared__ __bf16 VTs[64 * 64];
    __shared__ float  Ss[64 * 64];
    __shared__ __bf16 Ps[64 * 64];
    __shared__ float mrow[64], lrowS[64], resc[64];
    int b = blockIdx.z, h = blockIdx.y, q0 = blockIdx.x * QT;
    int tid = threadIdx.x, lane = tid & 31, wave = tid >> 5;
    int swm = wave >> 2, swn = wave & 3;
    int lr = lane & 15, lsel = (lane >> 4) * 16;
    int row = tid >> 2, seg = tid & 3;
    size_t baseQ = ((size_t)(b * H + h) * S + q0) * D;
    // async stage Q tile
    async_load_b128(lds_off(&Qs[row * 64 + seg * 16]),
                    qb + baseQ + (size_t)row * D + seg * 16);
    if (tid < QT) { mrow[tid] = -3.0e38f; lrowS[tid] = 0.f; }
    v8f oacc[2] = {};
    wait_async(0);
    __syncthreads();
    float scale = rsqrtf((float)D);
    for (int kt = 0; kt < S; kt += KT) {
        size_t baseK = ((size_t)(b * H + h) * S + kt) * D;
        // async stage K tile; V tile transposed via LDS scalar stores
        async_load_b128(lds_off(&Ks[row * 64 + seg * 16]),
                        kb + baseK + (size_t)row * D + seg * 16);
        {
            v16bf tv = *(const v16bf*)(vb + baseK + (size_t)row * D + seg * 16);
            for (int e = 0; e < 16; e++) VTs[(seg * 16 + e) * 64 + row] = tv[e];
        }
        wait_async(0);
        __syncthreads();
        // S = Q x K^T  (M=64 queries, N=64 keys, K=D)
        v8f sacc[2] = {};
        for (int ks = 0; ks < D; ks += 32) {
            v16bf afr[2], bfr;
            for (int i = 0; i < 2; i++)
                afr[i] = *(const v16bf*)&Qs[(swm * 32 + i * 16 + lr) * 64 + ks + lsel];
            bfr = *(const v16bf*)&Ks[(swn * 16 + lr) * 64 + ks + lsel];
            for (int i = 0; i < 2; i++)
                sacc[i] = __builtin_amdgcn_wmma_f32_16x16x32_bf16(
                    false, afr[i], false, bfr, (short)0, sacc[i], false, false);
        }
        const float* bp = biasT + (size_t)(b * H + h) * S + kt;
        for (int i = 0; i < 2; i++)
            for (int e = 0; e < 8; e++) {
                int r2 = swm * 32 + i * 16 + e + 8 * (lane >> 4);
                int col = swn * 16 + (lane & 15);
                Ss[r2 * 64 + col] = sacc[i][e] * scale + bp[col];
            }
        __syncthreads();
        if (tid < QT) {
            int r = tid;
            float mx = mrow[r];
            for (int c = 0; c < KT; c++) mx = fmaxf(mx, Ss[r * 64 + c]);
            float rsc = expf(mrow[r] - mx);
            float l = lrowS[r] * rsc;
            for (int c = 0; c < KT; c++) {
                float p = expf(Ss[r * 64 + c] - mx);
                Ps[r * 64 + c] = f2bf(p);
                l += p;
            }
            mrow[r] = mx; lrowS[r] = l; resc[r] = rsc;
        }
        __syncthreads();
        for (int i = 0; i < 2; i++)
            for (int e = 0; e < 8; e++) {
                int r2 = swm * 32 + i * 16 + e + 8 * (lane >> 4);
                oacc[i][e] *= resc[r2];
            }
        // O += P x V  (M=64 queries, N=64 dims, K=keys)
        for (int ks = 0; ks < KT; ks += 32) {
            v16bf afr[2], bfr;
            for (int i = 0; i < 2; i++)
                afr[i] = *(const v16bf*)&Ps[(swm * 32 + i * 16 + lr) * 64 + ks + lsel];
            bfr = *(const v16bf*)&VTs[(swn * 16 + lr) * 64 + ks + lsel];
            for (int i = 0; i < 2; i++)
                oacc[i] = __builtin_amdgcn_wmma_f32_16x16x32_bf16(
                    false, afr[i], false, bfr, (short)0, oacc[i], false, false);
        }
        __syncthreads();
    }
    for (int i = 0; i < 2; i++)
        for (int e = 0; e < 8; e++) {
            int r2 = swm * 32 + i * 16 + e + 8 * (lane >> 4);
            int col = swn * 16 + (lane & 15);
            float val = oacc[i][e] / lrowS[r2];
            ob[((size_t)(b * S) + q0 + r2) * Hd + h * D + col] = f2bf(val);
        }
}

// ---------------- host orchestration ----------------

extern "C" void kernel_launch(void* const* d_in, const int* in_sizes, int n_in,
                              void* d_out, int out_size, void* d_ws, size_t ws_size,
                              hipStream_t stream) {
    (void)in_sizes; (void)n_in; (void)out_size; (void)ws_size;
    const int B = 2, S = 1024, V = 32000, Hd = 1024, L = 4, H = 16, D = 64, I = 4096;
    const int BS = B * S;
    const int TOPK = S / 2;

    const int*   ids    = (const int*)d_in[0];
    const float* amask  = (const float*)d_in[1];
    const float* embed  = (const float*)d_in[2];
    const float* ln1_w  = (const float*)d_in[3];
    const float* ln1_b  = (const float*)d_in[4];
    const float* Wq     = (const float*)d_in[5];
    const float* Wk     = (const float*)d_in[6];
    const float* Wv     = (const float*)d_in[7];
    const float* Wo     = (const float*)d_in[8];
    const float* Wr     = (const float*)d_in[9];
    const float* ln2_w  = (const float*)d_in[10];
    const float* ln2_b  = (const float*)d_in[11];
    const float* W1     = (const float*)d_in[12];
    const float* b1     = (const float*)d_in[13];
    const float* W2     = (const float*)d_in[14];
    const float* b2     = (const float*)d_in[15];
    const float* norm_w = (const float*)d_in[16];
    const float* norm_b = (const float*)d_in[17];
    const float* lm     = (const float*)d_in[18];

    char* ws = (char*)d_ws;
    size_t off = 0;
    auto alloc = [&](size_t bytes) -> void* {
        void* p = ws + off;
        off = (off + bytes + 255) & ~(size_t)255;
        return p;
    };

    float*  x      = (float*)alloc((size_t)BS * Hd * 4);
    __bf16* hb     = (__bf16*)alloc((size_t)BS * Hd * 2);
    float*  qf     = (float*)alloc((size_t)BS * Hd * 4);
    float*  kf     = (float*)alloc((size_t)BS * Hd * 4);
    float*  vf     = (float*)alloc((size_t)BS * Hd * 4);
    __bf16* qbuf   = (__bf16*)alloc((size_t)BS * Hd * 2);
    __bf16* kbuf   = (__bf16*)alloc((size_t)BS * Hd * 2);
    __bf16* vbuf   = (__bf16*)alloc((size_t)BS * Hd * 2);
    __bf16* obuf   = (__bf16*)alloc((size_t)BS * Hd * 2);
    __bf16* mb     = (__bf16*)alloc((size_t)BS * I * 2);
    float*  router = (float*)alloc((size_t)B * H * S * 4);
    float*  kth    = (float*)alloc((size_t)B * H * 4);
    float*  biasT  = (float*)alloc((size_t)B * H * S * 4);
    __bf16* WqT    = (__bf16*)alloc((size_t)L * Hd * Hd * 2);
    __bf16* WkT    = (__bf16*)alloc((size_t)L * Hd * Hd * 2);
    __bf16* WvT    = (__bf16*)alloc((size_t)L * Hd * Hd * 2);
    __bf16* WoT    = (__bf16*)alloc((size_t)L * Hd * Hd * 2);
    __bf16* W1T    = (__bf16*)alloc((size_t)L * Hd * I * 2);
    __bf16* W2T    = (__bf16*)alloc((size_t)L * I * Hd * 2);
    __bf16* lmT    = (__bf16*)alloc((size_t)V * Hd * 2);

    for (int l = 0; l < L; l++) {
        int g1 = (Hd * Hd + 255) / 256;
        wtrans_kernel<<<g1, 256, 0, stream>>>(Wq + (size_t)l * Hd * Hd, WqT + (size_t)l * Hd * Hd, Hd, Hd);
        wtrans_kernel<<<g1, 256, 0, stream>>>(Wk + (size_t)l * Hd * Hd, WkT + (size_t)l * Hd * Hd, Hd, Hd);
        wtrans_kernel<<<g1, 256, 0, stream>>>(Wv + (size_t)l * Hd * Hd, WvT + (size_t)l * Hd * Hd, Hd, Hd);
        wtrans_kernel<<<g1, 256, 0, stream>>>(Wo + (size_t)l * Hd * Hd, WoT + (size_t)l * Hd * Hd, Hd, Hd);
        int g2 = (Hd * I + 255) / 256;
        wtrans_kernel<<<g2, 256, 0, stream>>>(W1 + (size_t)l * Hd * I, W1T + (size_t)l * Hd * I, Hd, I);
        wtrans_kernel<<<g2, 256, 0, stream>>>(W2 + (size_t)l * I * Hd, W2T + (size_t)l * I * Hd, I, Hd);
    }
    wtrans_kernel<<<(Hd * V + 255) / 256, 256, 0, stream>>>(lm, lmT, Hd, V);

    gather_kernel<<<(BS * Hd + 255) / 256, 256, 0, stream>>>(ids, embed, x, Hd, BS * Hd);

    dim3 gHd(Hd / 64, BS / 128), gI(I / 64, BS / 128), gV(V / 64, BS / 128);
    for (int l = 0; l < L; l++) {
        ln_kernel<<<BS, 256, 0, stream>>>(x, ln1_w + (size_t)l * Hd, ln1_b + (size_t)l * Hd, hb, Hd);
        gemm_bf16_kernel<0><<<gHd, 256, 0, stream>>>(hb, WqT + (size_t)l * Hd * Hd, nullptr, nullptr, qf, nullptr, BS, Hd, Hd);
        gemm_bf16_kernel<0><<<gHd, 256, 0, stream>>>(hb, WkT + (size_t)l * Hd * Hd, nullptr, nullptr, kf, nullptr, BS, Hd, Hd);
        gemm_bf16_kernel<0><<<gHd, 256, 0, stream>>>(hb, WvT + (size_t)l * Hd * Hd, nullptr, nullptr, vf, nullptr, BS, Hd, Hd);
        rope_kernel<<<(B * H * S * D + 255) / 256, 256, 0, stream>>>(qf, kf, vf, qbuf, kbuf, vbuf, B, H, S, D, Hd);
        router_kernel<<<(B * H * S + 255) / 256, 256, 0, stream>>>(hb, Wr + (size_t)l * Hd * H, router, B, H, S, Hd);
        kth_select_kernel<<<B * H, 256, 0, stream>>>(router, kth, S, TOPK);
        bias_kernel<<<(B * H * S + 255) / 256, 256, 0, stream>>>(router, kth, amask, biasT, B, H, S);
        flash_attn_kernel<<<dim3(S / 64, H, B), 256, 0, stream>>>(qbuf, kbuf, vbuf, biasT, obuf, B, H, S, D, Hd);
        gemm_bf16_kernel<0><<<gHd, 256, 0, stream>>>(obuf, WoT + (size_t)l * Hd * Hd, nullptr, x, x, nullptr, BS, Hd, Hd);
        ln_kernel<<<BS, 256, 0, stream>>>(x, ln2_w + (size_t)l * Hd, ln2_b + (size_t)l * Hd, hb, Hd);
        gemm_bf16_kernel<1><<<gI, 256, 0, stream>>>(hb, W1T + (size_t)l * Hd * I, b1 + (size_t)l * I, nullptr, nullptr, mb, BS, I, Hd);
        gemm_bf16_kernel<0><<<gHd, 256, 0, stream>>>(mb, W2T + (size_t)l * I * Hd, b2 + (size_t)l * Hd, x, x, nullptr, BS, Hd, I);
    }
    ln_kernel<<<BS, 256, 0, stream>>>(x, norm_w, norm_b, hb, Hd);
    gemm_bf16_kernel<0><<<gV, 256, 0, stream>>>(hb, lmT, nullptr, nullptr, (float*)d_out, nullptr, BS, V, Hd);
}
